// CoAttentionPara_38156489457740
// MI455X (gfx1250) — matrix operand order repaired and checked by reading, not compiled, not run on hardware
//
#include <hip/hip_runtime.h>
#include <hip/hip_bf16.h>
#include <math.h>

typedef __bf16 bf16_t;
typedef __attribute__((ext_vector_type(16))) __bf16 v16bf;
typedef __attribute__((ext_vector_type(8)))  __bf16 bf16x8;
typedef __attribute__((ext_vector_type(8)))  float  v8f;

// Problem constants
#define BB 32
#define SS 512
#define RR 196
#define RP 224      // R padded to multiple of 32
#define EE 1024
#define KK 512

// GEMM tiling
#define BM 64
#define BN 128
#define BK 32
#define LDA_S 40    // 32 + 8 pad (80B rows, 16B-aligned)
#define LDB_S 136   // 128 + 8 pad (272B rows, 16B-aligned)

// CDNA5 async global->LDS copy, 16 bytes per lane (ASYNCcnt-tracked).
__device__ __forceinline__ void async_copy_b128(unsigned lds_off,
                                                const bf16_t* gptr) {
    asm volatile("global_load_async_to_lds_b128 %0, %1, off"
                 :: "v"(lds_off), "v"(gptr)
                 : "memory");
}

// ---------------------------------------------------------------------------
// Generic bf16 WMMA GEMM: C = A[MxKd] * B[KdxN], f32 accumulate.
// Epilogues:
//  0: store bf16 to O0
//  1: store f32 to O0f and bf16 to O1
//  2: store bf16 to O0 and bf16-transposed to O1  (O1[n*ldo1+m])
//  3: fused attention: h=tanh(aux[m,n]+acc); atomicAdd(logit[m], sum_n h*w[n])
// ---------------------------------------------------------------------------
template <int EPI>
__global__ __launch_bounds__(256) void gemm_bf16_k(
    const bf16_t* __restrict__ A, long long sA, int lda,
    const bf16_t* __restrict__ Bm, long long sB, int ldb,
    int M, int N, int Kt,
    float*  __restrict__ O0f, long long sO0, int ldo0,
    bf16_t* __restrict__ O0b,
    bf16_t* __restrict__ O1b, long long sO1, int ldo1,
    const float* __restrict__ aux, long long sAux, int ldAux,
    const float* __restrict__ wvec,
    float* __restrict__ logit, long long sL, int Mreal)
{
    __shared__ bf16_t As[2 * BM * LDA_S];
    __shared__ bf16_t Bs[2 * BK * LDB_S];

    const int b    = blockIdx.z;
    const int bm0  = blockIdx.y * BM;
    const int bn0  = blockIdx.x * BN;
    const int tid  = threadIdx.x;
    const int lane = tid & 31;
    const int wave = tid >> 5;
    const int wm0  = (wave >> 2) * 32;   // 2 waves along M
    const int wn0  = (wave & 3) * 32;    // 4 waves along N
    const int half = lane >> 4;
    const int l15  = lane & 15;

    const bf16_t* Ab = A  + (long long)b * sA;
    const bf16_t* Bb = Bm + (long long)b * sB;

    v8f acc[2][2];
#pragma unroll
    for (int i = 0; i < 2; ++i)
#pragma unroll
        for (int j = 0; j < 2; ++j)
#pragma unroll
            for (int v = 0; v < 8; ++v) acc[i][j][v] = 0.0f;

    // cooperative staging coordinates (uniform issue count: 3 asyncs/thread)
    const int ar = tid >> 2;            // 0..63
    const int ac = (tid & 3) * 8;       // 0,8,16,24
    const int br = tid >> 3;            // 0..31
    const int bc = (tid & 7) * 16;      // 0..112
    // clamped addresses: OOB rows/cols fetch duplicate (finite) data; the
    // resulting garbage output rows/cols are masked by the store guards.
    const int gm = bm0 + ar;
    const int gn = bn0 + bc;
    const int gmc = (gm < M) ? gm : (M - 1);
    const int gnc = (gn < N) ? gn : (N - 16);

    const unsigned asA = (unsigned)(size_t)(&As[0]) + (unsigned)((ar * LDA_S + ac) * 2);
    const unsigned asB = (unsigned)(size_t)(&Bs[0]) + (unsigned)((br * LDB_S + bc) * 2);

    auto stage = [&](int buf, int k0) {
        async_copy_b128(asA + (unsigned)(buf * BM * LDA_S * 2),
                        Ab + (long long)gmc * lda + (k0 + ac));
        const bf16_t* bp = Bb + (long long)(k0 + br) * ldb + gnc;
        const unsigned bo = asB + (unsigned)(buf * BK * LDB_S * 2);
        async_copy_b128(bo,      bp);
        async_copy_b128(bo + 16, bp + 8);
    };

    stage(0, 0);                         // prologue: tile 0 in flight

    for (int kb = 0; kb < Kt; ++kb) {
        if (kb + 1 < Kt) {
            stage((kb + 1) & 1, (kb + 1) * BK);
            // 3 asyncs of the next stage stay in flight; current stage done.
            asm volatile("s_wait_asynccnt 0x3" ::: "memory");
        } else {
            asm volatile("s_wait_asynccnt 0x0" ::: "memory");
        }
        __syncthreads();

        const bf16_t* Asb = &As[(kb & 1) * BM * LDA_S];
        const bf16_t* Bsb = &Bs[(kb & 1) * BK * LDB_S];

        // ---- fragments per ISA 7.12.2 layouts ----
        v16bf afrag[2], bfrag[2];
#pragma unroll
        for (int i = 0; i < 2; ++i) {
            const int m = wm0 + i * 16 + l15;
            bf16x8 lo = *(const bf16x8*)(&Asb[m * LDA_S + half * 8]);
            bf16x8 hi = *(const bf16x8*)(&Asb[m * LDA_S + 16 + half * 8]);
            afrag[i] = __builtin_shufflevector(lo, hi,
                0,1,2,3,4,5,6,7,8,9,10,11,12,13,14,15);
        }
#pragma unroll
        for (int j = 0; j < 2; ++j) {
            const int n = wn0 + j * 16 + l15;
#pragma unroll
            for (int s = 0; s < 16; ++s) {
                const int k = (s < 8) ? (half * 8 + s) : (16 + half * 8 + (s - 8));
                bfrag[j][s] = Bsb[k * LDB_S + n];
            }
        }
#pragma unroll
        for (int i = 0; i < 2; ++i)
#pragma unroll
            for (int j = 0; j < 2; ++j)
                acc[i][j] = __builtin_amdgcn_wmma_f32_16x16x32_bf16(
                    false, afrag[i], false, bfrag[j],
                    (short)0, acc[i][j], false, false);
        __syncthreads();
    }

    // ---- epilogue ----
#pragma unroll
    for (int i = 0; i < 2; ++i) {
#pragma unroll
        for (int j = 0; j < 2; ++j) {
            const int mBase = bm0 + wm0 + i * 16 + half * 8;
            const int n     = bn0 + wn0 + j * 16 + l15;
#pragma unroll
            for (int v = 0; v < 8; ++v) {
                const int m = mBase + v;
                const float x = acc[i][j][v];
                if (EPI == 3) {
                    float p = 0.0f;
                    if (m < M && n < N) {
                        const float base =
                            aux[(long long)b * sAux + (long long)m * ldAux + n];
                        p = tanhf(base + x) * wvec[n];
                    }
                    // reduce over the 16 lanes holding row m (N dimension)
                    p += __shfl_xor(p, 1);
                    p += __shfl_xor(p, 2);
                    p += __shfl_xor(p, 4);
                    p += __shfl_xor(p, 8);
                    if (l15 == 0 && m < Mreal)
                        atomicAdd(logit + (long long)b * sL + m, p);
                } else if (m < M && n < N) {
                    const long long o = (long long)b * sO0 + (long long)m * ldo0 + n;
                    if (EPI == 0) {
                        O0b[o] = (__bf16)x;
                    } else if (EPI == 1) {
                        O0f[o] = x;
                        O1b[(long long)b * sO1 + (long long)m * ldo1 + n] = (__bf16)x;
                    } else { // EPI == 2
                        O0b[o] = (__bf16)x;
                        O1b[(long long)b * sO1 + (long long)n * ldo1 + m] = (__bf16)x;
                    }
                }
            }
        }
    }
}

// ---------------------------------------------------------------------------
__global__ __launch_bounds__(256) void cvt_bf16_k(const float* __restrict__ src,
                                                  bf16_t* __restrict__ dst,
                                                  long long n) {
    long long i = (long long)blockIdx.x * 256 + threadIdx.x;
    if (i < n) dst[i] = (__bf16)src[i];
}

// Itr[b][e][rp] = I[b][r][e], zero-padded r in [RR,RP)
__global__ __launch_bounds__(256) void transpose_I_k(const float* __restrict__ I,
                                                     bf16_t* __restrict__ Itr) {
    long long i = (long long)blockIdx.x * 256 + threadIdx.x;
    const long long total = (long long)BB * EE * RP;
    if (i >= total) return;
    int r = (int)(i % RP);
    long long be = i / RP;
    int e = (int)(be % EE);
    int b = (int)(be / EE);
    float v = (r < RR) ? I[((long long)b * RR + r) * EE + e] : 0.0f;
    Itr[i] = (__bf16)v;
}

__global__ __launch_bounds__(256) void zero_f32_k(float* __restrict__ p, long long n) {
    long long i = (long long)blockIdx.x * 256 + threadIdx.x;
    if (i < n) p[i] = 0.0f;
}

// one block per batch, softmax over n elements (stride = per-batch stride)
__global__ __launch_bounds__(256) void softmax_k(const float* __restrict__ in,
                                                 float* __restrict__ out,
                                                 int n, int stride) {
    __shared__ float sh[256];
    const int b = blockIdx.x;
    const int tid = threadIdx.x;
    const float* x = in + (long long)b * stride;
    float* y = out + (long long)b * stride;

    float mx = -INFINITY;
    for (int i = tid; i < n; i += 256) mx = fmaxf(mx, x[i]);
    sh[tid] = mx; __syncthreads();
    for (int s = 128; s > 0; s >>= 1) {
        if (tid < s) sh[tid] = fmaxf(sh[tid], sh[tid + s]);
        __syncthreads();
    }
    mx = sh[0]; __syncthreads();

    float sum = 0.0f;
    for (int i = tid; i < n; i += 256) sum += expf(x[i] - mx);
    sh[tid] = sum; __syncthreads();
    for (int s = 128; s > 0; s >>= 1) {
        if (tid < s) sh[tid] += sh[tid + s];
        __syncthreads();
    }
    sum = sh[0];

    const float inv = 1.0f / sum;
    for (int i = tid; i < n; i += 256) y[i] = expf(x[i] - mx) * inv;
}

// out[b,m,e] = w[b*wstride + m] * src[b,m,e]  (vectorized float4 over e)
__global__ __launch_bounds__(256) void scale_k(const float* __restrict__ src,
                                               const float* __restrict__ w,
                                               float* __restrict__ out,
                                               long long total4, int Mrows, int wstride) {
    long long i = (long long)blockIdx.x * 256 + threadIdx.x;
    if (i >= total4) return;
    long long f = i * 4;
    long long me = f / EE;          // b*Mrows + m
    int m = (int)(me % Mrows);
    int b = (int)(me / Mrows);
    float a = w[(long long)b * wstride + m];
    float4 v = *(const float4*)(src + f);
    v.x *= a; v.y *= a; v.z *= a; v.w *= a;
    *(float4*)(out + f) = v;
}

// ---------------------------------------------------------------------------
extern "C" void kernel_launch(void* const* d_in, const int* in_sizes, int n_in,
                              void* d_out, int out_size, void* d_ws, size_t ws_size,
                              hipStream_t stream) {
    (void)in_sizes; (void)n_in; (void)out_size; (void)ws_size;
    const float* T   = (const float*)d_in[0];   // [B,S,E]
    const float* I   = (const float*)d_in[1];   // [B,R,E]
    const float* Wb  = (const float*)d_in[2];   // [E,E]
    const float* Wq  = (const float*)d_in[3];   // [E,K]
    const float* Wv  = (const float*)d_in[4];   // [E,K]
    const float* Whv = (const float*)d_in[5];   // [K,1]
    const float* Whq = (const float*)d_in[7];   // [K,1]
    float* out = (float*)d_out;

    char* ws = (char*)d_ws;
    size_t off = 0;
    auto carve = [&](size_t bytes) -> char* {
        char* p = ws + off;
        off = (off + bytes + 255) & ~(size_t)255;
        return p;
    };

    bf16_t* Tb    = (bf16_t*)carve((size_t)BB * SS * EE * 2);
    bf16_t* Ib    = (bf16_t*)carve((size_t)BB * RR * EE * 2);
    bf16_t* Itr   = (bf16_t*)carve((size_t)BB * EE * RP * 2);
    bf16_t* Wbb   = (bf16_t*)carve((size_t)EE * EE * 2);
    bf16_t* Wqb   = (bf16_t*)carve((size_t)EE * KK * 2);
    bf16_t* Wvb   = (bf16_t*)carve((size_t)EE * KK * 2);
    bf16_t* TWbb  = (bf16_t*)carve((size_t)BB * SS * EE * 2);
    bf16_t* affib = (bf16_t*)carve((size_t)BB * SS * RP * 2);
    bf16_t* affiT = (bf16_t*)carve((size_t)BB * RP * SS * 2);
    float*  tWqf  = (float*) carve((size_t)BB * SS * KK * 4);
    bf16_t* tWqb  = (bf16_t*)carve((size_t)BB * SS * KK * 2);
    float*  iWvf  = (float*) carve((size_t)BB * RP * KK * 4);
    bf16_t* iWvb  = (bf16_t*)carve((size_t)BB * RP * KK * 2);
    float*  avL   = (float*) carve((size_t)BB * RP * 4);
    float*  aqL   = (float*) carve((size_t)BB * SS * 4);
    float*  avP   = (float*) carve((size_t)BB * RP * 4);
    float*  aqP   = (float*) carve((size_t)BB * SS * 4);

    auto blocks = [](long long n) { return (unsigned)((n + 255) / 256); };

    // ---- precision conversions ----
    cvt_bf16_k<<<blocks((long long)BB * SS * EE), 256, 0, stream>>>(T, Tb, (long long)BB * SS * EE);
    cvt_bf16_k<<<blocks((long long)BB * RR * EE), 256, 0, stream>>>(I, Ib, (long long)BB * RR * EE);
    cvt_bf16_k<<<blocks((long long)EE * EE), 256, 0, stream>>>(Wb, Wbb, (long long)EE * EE);
    cvt_bf16_k<<<blocks((long long)EE * KK), 256, 0, stream>>>(Wq, Wqb, (long long)EE * KK);
    cvt_bf16_k<<<blocks((long long)EE * KK), 256, 0, stream>>>(Wv, Wvb, (long long)EE * KK);
    transpose_I_k<<<blocks((long long)BB * EE * RP), 256, 0, stream>>>(I, Itr);
    zero_f32_k<<<blocks((long long)BB * RP), 256, 0, stream>>>(avL, (long long)BB * RP);
    zero_f32_k<<<blocks((long long)BB * SS), 256, 0, stream>>>(aqL, (long long)BB * SS);

    // ---- GEMM 1: TWb[S,E] = T[S,E] * Wb[E,E] ----
    gemm_bf16_k<0><<<dim3(EE / BN, SS / BM, BB), 256, 0, stream>>>(
        Tb, (long long)SS * EE, EE, Wbb, 0, EE,
        SS, EE, EE / BK,
        nullptr, (long long)SS * EE, EE, TWbb,
        nullptr, 0, 0, nullptr, 0, 0, nullptr, nullptr, 0, 0);

    // ---- GEMM 2: affi[S,RP] = TWb[S,E] * Itr[E,RP]; also store affi^T ----
    gemm_bf16_k<2><<<dim3((RP + BN - 1) / BN, SS / BM, BB), 256, 0, stream>>>(
        TWbb, (long long)SS * EE, EE, Itr, (long long)EE * RP, RP,
        SS, RP, EE / BK,
        nullptr, (long long)SS * RP, RP, affib,
        affiT, (long long)RP * SS, SS, nullptr, 0, 0, nullptr, nullptr, 0, 0);

    // ---- GEMM 3: tWq[S,K] = T[S,E] * Wq[E,K]  (f32 + bf16) ----
    gemm_bf16_k<1><<<dim3(KK / BN, SS / BM, BB), 256, 0, stream>>>(
        Tb, (long long)SS * EE, EE, Wqb, 0, KK,
        SS, KK, EE / BK,
        tWqf, (long long)SS * KK, KK, nullptr,
        tWqb, (long long)SS * KK, KK, nullptr, 0, 0, nullptr, nullptr, 0, 0);

    // ---- GEMM 4: iWv[R,K] = I[R,E] * Wv[E,K]  (f32 + bf16, RP-strided) ----
    gemm_bf16_k<1><<<dim3(KK / BN, (RR + BM - 1) / BM, BB), 256, 0, stream>>>(
        Ib, (long long)RR * EE, EE, Wvb, 0, KK,
        RR, KK, EE / BK,
        iWvf, (long long)RP * KK, KK, nullptr,
        iWvb, (long long)RP * KK, KK, nullptr, 0, 0, nullptr, nullptr, 0, 0);

    // ---- GEMM 5 (fused): av logits: tanh(iWv + affi^T * tWq) . Whv ----
    gemm_bf16_k<3><<<dim3(KK / BN, (RR + BM - 1) / BM, BB), 256, 0, stream>>>(
        affiT, (long long)RP * SS, SS, tWqb, (long long)SS * KK, KK,
        RR, KK, SS / BK,
        avL, RP, 0, nullptr,
        nullptr, 0, 0,
        iWvf, (long long)RP * KK, KK, Whv, avL, RP, RR);

    // ---- GEMM 6 (fused): aq logits: tanh(tWq + affi * iWv) . Whq ----
    gemm_bf16_k<3><<<dim3(KK / BN, SS / BM, BB), 256, 0, stream>>>(
        affib, (long long)SS * RP, RP, iWvb, (long long)RP * KK, KK,
        SS, KK, RP / BK,
        aqL, SS, 0, nullptr,
        nullptr, 0, 0,
        tWqf, (long long)SS * KK, KK, Whq, aqL, SS, SS);

    // ---- softmaxes (scalar biases are shift-invariant, dropped) ----
    softmax_k<<<BB, 256, 0, stream>>>(avL, avP, RR, RP);
    softmax_k<<<BB, 256, 0, stream>>>(aqL, aqP, SS, SS);

    // ---- scale features ----
    const long long t4 = (long long)BB * SS * EE / 4;
    const long long i4 = (long long)BB * RR * EE / 4;
    scale_k<<<blocks(t4), 256, 0, stream>>>(T, aqP, out, t4, SS, SS);
    scale_k<<<blocks(i4), 256, 0, stream>>>(I, avP, out + (long long)BB * SS * EE,
                                            i4, RR, RP);
}
